// BinaryLinear_91113436217704
// MI455X (gfx1250) — matrix operand order repaired and checked by reading, not compiled, not run on hardware
//
#include <hip/hip_runtime.h>
#include <cstdint>
#include <cstddef>

// ---- problem constants (match reference) ----
#define T_DIM 8192   // tokens (M)
#define K_DIM 4096   // in_features (K)
#define O_DIM 4096   // out_features (N)

// ---- tiling ----
#define BM 128       // block tile M
#define BN 128       // block tile N
#define KT 64        // K step per LDS stage (2 WMMA K-steps of 32)
#define LDW 72       // padded LDS row stride in halfs (64 + 8): 144B rows, 16B-aligned frags

typedef __attribute__((ext_vector_type(16))) _Float16 v16h;
typedef __attribute__((ext_vector_type(8)))  _Float16 v8h;
typedef __attribute__((ext_vector_type(8)))  float    v8f;

union V16 { v16h v; v8h h[2]; };

// binarize two packed f32 sign bits -> two packed f16 (+1.0h / -1.0h)
// v_perm_b32 gathers [w1.byte3 -> byte3, w0.byte3 -> byte1], mask keeps the
// f16 sign-bit positions, OR injects the 1.0h magnitude. 3 VALU ops total.
__device__ __forceinline__ uint32_t binpack2(uint32_t w0, uint32_t w1) {
    uint32_t signs = __builtin_amdgcn_perm(w1, w0, 0x07040300u);
    return 0x3C003C00u | (signs & 0x80008000u);
}

__global__ __launch_bounds__(256, 2)
void binlin_wmma_f16(const float* __restrict__ X,
                     const float* __restrict__ W,
                     const float* __restrict__ scale,
                     float* __restrict__ Y)
{
    // double-buffered f16 tiles: A = x tile (BM x KT), B = binarized w tile (BN x KT)
    __shared__ _Float16 As[2][BM * LDW];   // 2 * 18KB
    __shared__ _Float16 Bs[2][BN * LDW];   // 2 * 18KB  (72KB total)

    const int tid  = threadIdx.x;
    const int lane = tid & 31;
    const int wav  = tid >> 5;      // 8 waves
    const int lh   = lane & 15;     // lane within half-wave
    const int sel  = lane >> 4;     // 0: lanes 0-15, 1: lanes 16-31

    const int nBase = blockIdx.x * BN;   // out_features
    const int mBase = blockIdx.y * BM;   // tokens

    // wave -> 64x32 sub-tile of the 128x128 block tile (2 waves in M, 4 in N)
    const int wm = (wav & 1) * 64;
    const int wn = (wav >> 1) * 32;

    // cooperative global->LDS mapping: each thread loads 32 consecutive values of one row
    const int lr = tid >> 1;           // 0..127 row within tile
    const int lc = (tid & 1) * 32;     // 0 or 32 column within K-step
    const float* gA = X + (size_t)(mBase + lr) * K_DIM + lc;
    const uint4* gB = (const uint4*)(W + (size_t)(nBase + lr) * K_DIM + lc);

    float4 ra[8];
    uint4  rbu[8];

    auto g_load = [&](int kt) {
        const float4* pa = (const float4*)(gA + kt);
        const uint4*  pb = gB + (kt >> 2);
        #pragma unroll
        for (int i = 0; i < 8; ++i) { ra[i] = pa[i]; rbu[i] = pb[i]; }
    };

    auto lds_store = [&](int buf) {
        // A: convert f32 -> f16 (v_cvt_pk_f16_f32), 4x ds_store_b128
        _Float16* da = &As[buf][lr * LDW + lc];
        #pragma unroll
        for (int j = 0; j < 4; ++j) {
            const float* f = (const float*)&ra[2 * j];  // 8 floats
            v8h h;
            #pragma unroll
            for (int i = 0; i < 8; ++i) h[i] = (_Float16)f[i];
            *(v8h*)(da + 8 * j) = h;
        }
        // B: binarize sign(w) -> +/-1.0h, 3 VALU per packed dword via v_perm_b32
        uint32_t pk[16];
        #pragma unroll
        for (int i = 0; i < 8; ++i) {
            pk[2*i]     = binpack2(rbu[i].x, rbu[i].y);
            pk[2*i + 1] = binpack2(rbu[i].z, rbu[i].w);
        }
        _Float16* db = &Bs[buf][lr * LDW + lc];
        #pragma unroll
        for (int j = 0; j < 4; ++j)
            *(uint4*)(db + 8 * j) = *(uint4*)(pk + 4 * j);
    };

    v8f acc[4][2];
    #pragma unroll
    for (int mi = 0; mi < 4; ++mi)
        #pragma unroll
        for (int ni = 0; ni < 2; ++ni)
            #pragma unroll
            for (int i = 0; i < 8; ++i)
                acc[mi][ni][i] = 0.0f;

    // prologue: fill buffer 0
    g_load(0);
    lds_store(0);

    const int kIters = K_DIM / KT;   // 64
    int buf = 0;
    for (int k = 0; k < kIters; ++k) {
        __syncthreads();
        if (k + 1 < kIters) g_load((k + 1) * KT);  // global prefetch overlaps compute

        // ---- 2 WMMA K-steps (K=32 each) out of this LDS stage ----
        #pragma unroll
        for (int ks = 0; ks < 2; ++ks) {
            // A 16x32 f16 frag: lane lh = row; halfs 0..7 = K[sel*8..+7],
            // halfs 8..15 = K[sel*8+16..+23] -> two 16B contiguous runs.
            V16 af[4];
            #pragma unroll
            for (int mi = 0; mi < 4; ++mi) {
                const _Float16* p = &As[buf][(wm + mi * 16 + lh) * LDW + ks * 32 + sel * 8];
                af[mi].h[0] = *(const v8h*)(p);
                af[mi].h[1] = *(const v8h*)(p + 16);
            }
            // B 32x16 f16 frag: lane lh = col N (== weight row o);
            // halfs 0..7 = K[sel*8..+7], halfs 8..15 = K[sel*8+16..+23] -> two 16B runs.
            V16 bf2[2];
            #pragma unroll
            for (int ni = 0; ni < 2; ++ni) {
                const _Float16* p = &Bs[buf][(wn + ni * 16 + lh) * LDW + ks * 32 + sel * 8];
                bf2[ni].h[0] = *(const v8h*)(p);
                bf2[ni].h[1] = *(const v8h*)(p + 16);
            }
            #pragma unroll
            for (int mi = 0; mi < 4; ++mi)
                #pragma unroll
                for (int ni = 0; ni < 2; ++ni)
                    acc[mi][ni] = __builtin_amdgcn_wmma_f32_16x16x32_f16(
                        /*neg_a=*/false, af[mi].v,
                        /*neg_b=*/false, bf2[ni].v,
                        /*c_mod=*/(short)0, acc[mi][ni],
                        /*reuse_a=*/false, /*reuse_b=*/false);
        }

        if (k + 1 < kIters) lds_store(buf ^ 1);  // write other buffer: one barrier per stage
        buf ^= 1;
    }

    // ---- epilogue: scale in f32, write C (VGPR v -> M = v + sel*8, N = lh) ----
    const float s = scale[0];
    #pragma unroll
    for (int mi = 0; mi < 4; ++mi) {
        #pragma unroll
        for (int ni = 0; ni < 2; ++ni) {
            const int n  = nBase + wn + ni * 16 + lh;
            const int m0 = mBase + wm + mi * 16 + sel * 8;
            #pragma unroll
            for (int v = 0; v < 8; ++v)
                Y[(size_t)(m0 + v) * O_DIM + n] = acc[mi][ni][v] * s;
        }
    }
}

extern "C" void kernel_launch(void* const* d_in, const int* in_sizes, int n_in,
                              void* d_out, int out_size, void* d_ws, size_t ws_size,
                              hipStream_t stream) {
    (void)in_sizes; (void)n_in; (void)out_size; (void)d_ws; (void)ws_size;
    const float* x     = (const float*)d_in[0];   // (8192, 4096) f32
    const float* w     = (const float*)d_in[1];   // (4096, 4096) f32
    const float* scale = (const float*)d_in[2];   // scalar f32
    float*       y     = (float*)d_out;           // (8192, 4096) f32

    dim3 grid(O_DIM / BN, T_DIM / BM);  // (32, 64): N-fastest keeps 64MB weights hot in 192MB L2
    dim3 block(256);
    binlin_wmma_f16<<<grid, block, 0, stream>>>(x, w, scale, y);
}
